// TorchInstanceProcessor_43465069035685
// MI455X (gfx1250) — matrix operand order repaired and checked by reading, not compiled, not run on hardware
//
#include <hip/hip_runtime.h>
#include <hip/hip_bf16.h>

typedef float    v4f  __attribute__((ext_vector_type(4)));
typedef float    v8f  __attribute__((ext_vector_type(8)));
typedef _Float16 v16h __attribute__((ext_vector_type(16)));

constexpr int S    = 2048;
constexpr int D    = 1024;
constexpr int R    = 16;
constexpr int NL   = 64;
constexpr int OUTD = 2 * D + R;   // 2064

// ---------------------------------------------------------------------------
// Span mean: sum up to 31 rows of 1024 f32, scale by 1/len, NT-store.
// One 256-thread block per pair; each lane owns a float4 column chunk.
// 4 independent accumulators -> 4 global_load_b128 in flight per lane.
// ---------------------------------------------------------------------------
__device__ __forceinline__ void span_mean_store(const float* __restrict__ base,
                                                int s, int e, int col,
                                                float* __restrict__ dst) {
  v4f a0 = {0.f, 0.f, 0.f, 0.f}, a1 = a0, a2 = a0, a3 = a0;
  int t = s;
  for (; t + 4 <= e; t += 4) {
    const float* r0 = base + (size_t)t * D + col;
    a0 += *(const v4f*)(r0);
    a1 += *(const v4f*)(r0 + D);
    a2 += *(const v4f*)(r0 + 2 * D);
    a3 += *(const v4f*)(r0 + 3 * D);
  }
  for (; t < e; ++t)
    a0 += *(const v4f*)(base + (size_t)t * D + col);
  v4f acc = (a0 + a1) + (a2 + a3);
  const float inv = 1.0f / (float)(e - s);
  acc *= inv;
  // Output streams once; keep tokvecs resident in the 192MB L2 instead.
  __builtin_nontemporal_store(acc, (v4f*)dst);
}

__global__ void span_mean_kernel(const float* __restrict__ tok,
                                 const int* __restrict__ bidx,
                                 const int* __restrict__ e1s,
                                 const int* __restrict__ e1e,
                                 const int* __restrict__ e2s,
                                 const int* __restrict__ e2e,
                                 float* __restrict__ out) {
  const int p   = blockIdx.x;
  const int col = (int)threadIdx.x * 4;       // 256 lanes * 4 = 1024 cols
  const int b   = bidx[p];
  const int s1  = e1s[p], e1 = e1e[p];
  const int s2  = e2s[p], e2 = e2e[p];
  const float* base = tok + (size_t)b * S * D;

  // Warm the head of span 2 while span 1 streams (global_prefetch_b8).
#pragma unroll
  for (int t = 0; t < 4; ++t)
    if (s2 + t < e2)
      __builtin_prefetch(base + (size_t)(s2 + t) * D + col, 0, 3);

  float* orow = out + (size_t)p * OUTD + col;
  span_mean_store(base, s1, e1, col, orow);
  span_mean_store(base, s2, e2, col, orow + D);
}

// ---------------------------------------------------------------------------
// rel = onehot(rel_idx) [P x 64] x rel_table [64 x 16]  via WMMA f16->f32.
// All values are exactly 0/1 so f16 is exact. One wave handles 16 pairs:
// two V_WMMA_F32_16X16X32_F16 steps over K = 0..63, C chained.
//
// Layouts used here (wave32, contiguous k per lane half):
//  A 16x32 f16:  lanes 0-15 (M=lane):  a[0..15]=K 0..15
//                lanes 16-31 (M=lane-16): a[0..15]=K 16..31
//  B 32x16 f16:  lanes 0-15 (N=lane):  b[k]=B[K=k,   N]
//                lanes 16-31 (N=lane-16): b[k]=B[K=16+k, N]
//  D 16x16 f32:  VGPR v, lanes 0-15 -> M=v, N=lane; lanes 16-31 -> M=8+v.
// ---------------------------------------------------------------------------
__global__ void rel_wmma_kernel(const int* __restrict__ rel_idx,
                                const float* __restrict__ rel_table,
                                float* __restrict__ out) {
  const int lane = (int)threadIdx.x;      // 0..31, one wave per block
  const int m    = lane & 15;             // pair-in-tile / N column
  const int half = lane >> 4;             // 0 or 1
  const int p0   = (int)blockIdx.x * 16;
  const int r    = rel_idx[p0 + m];       // this M-row's one-hot index

  v8f c = {};
#pragma unroll
  for (int kb = 0; kb < NL; kb += 32) {
    v16h a, b;
    const int aoff = kb + half * 16;
#pragma unroll
    for (int j = 0; j < 8; ++j) {
      a[j]     = (aoff + j      == r) ? (_Float16)1.0f : (_Float16)0.0f;
      a[8 + j] = (aoff + 8 + j  == r) ? (_Float16)1.0f : (_Float16)0.0f;
    }
    const int boff = kb + half * 16;
#pragma unroll
    for (int k = 0; k < 16; ++k)
      b[k] = (_Float16)rel_table[(size_t)(boff + k) * R + m];
    c = __builtin_amdgcn_wmma_f32_16x16x32_f16(
        /*neg_a=*/false, a, /*neg_b=*/false, b,
        /*c_mod=*/(short)0, c, /*reuse_a=*/false, /*reuse_b=*/false);
  }

#pragma unroll
  for (int v = 0; v < 8; ++v) {
    const int row = half * 8 + v;                       // M
    __builtin_nontemporal_store(
        c[v], out + (size_t)(p0 + row) * OUTD + 2 * D + m);
  }
}

// ---------------------------------------------------------------------------
extern "C" void kernel_launch(void* const* d_in, const int* in_sizes, int n_in,
                              void* d_out, int out_size, void* d_ws, size_t ws_size,
                              hipStream_t stream) {
  const float* tok  = (const float*)d_in[0];
  const int*   bidx = (const int*)d_in[1];
  const int*   e1s  = (const int*)d_in[2];
  const int*   e1e  = (const int*)d_in[3];
  const int*   e2s  = (const int*)d_in[4];
  const int*   e2e  = (const int*)d_in[5];
  const int*   ridx = (const int*)d_in[6];
  const float* rtab = (const float*)d_in[7];
  float*       out  = (float*)d_out;

  const int P = in_sizes[1];   // 8192 pairs

  span_mean_kernel<<<P, 256, 0, stream>>>(tok, bidx, e1s, e1e, e2s, e2e, out);
  rel_wmma_kernel<<<P / 16, 32, 0, stream>>>(ridx, rtab, out);
}
